// Non_local_84799834292404
// MI455X (gfx1250) — matrix operand order repaired and checked by reading, not compiled, not run on hardware
//
#include <hip/hip_runtime.h>

// Non-local block (attention) for B=32, C=512, Ci=256, N=1024 on gfx1250.
// All heavy GEMMs run on v_wmma_f32_16x16x32_f16 (wave32 WMMA), f16 data /
// f32 accumulation. The O(N^2) attention matrix is tiled through LDS and
// never written to HBM. phi staging in kernel 2 uses CDNA5 async
// global->LDS copies (ASYNCcnt). Workspace: 4 x 16MB f16 (gT,thetaT,phiT,YT).

#define Bq   32
#define Cc   512
#define CI   256
#define Nn   1024
#define EPSC 1e-5f

typedef _Float16 h8   __attribute__((ext_vector_type(8)));
typedef _Float16 v16h __attribute__((ext_vector_type(16)));
typedef float    v8f  __attribute__((ext_vector_type(8)));

__device__ __forceinline__ v16h cat16(h8 lo, h8 hi) {
  return __builtin_shufflevector(lo, hi, 0,1,2,3,4,5,6,7,8,9,10,11,12,13,14,15);
}
__device__ __forceinline__ v8f zero8() {
  v8f z;
#pragma unroll
  for (int i = 0; i < 8; ++i) z[i] = 0.0f;
  return z;
}
#define WMMA(A,Bv,Cv) __builtin_amdgcn_wmma_f32_16x16x32_f16(false,(A),false,(Bv),(short)0,(Cv),false,false)

// ---------------------------------------------------------------------------
// Kernel 1: projections P = W(proj) * X[b] + bias, stored transposed f16:
//   PT[b][n][ci], proj 0->g, 1->theta, 2->phi.
// grid (8 nblk, 32 b, 3 proj), block 256.
// ---------------------------------------------------------------------------
__global__ void __launch_bounds__(256)
nl_proj_kernel(const float* __restrict__ x,
               const float* __restrict__ w_g, const float* __restrict__ b_g,
               const float* __restrict__ w_t, const float* __restrict__ b_t,
               const float* __restrict__ w_p, const float* __restrict__ b_p,
               _Float16* __restrict__ ws) {
  const int nblk = blockIdx.x, b = blockIdx.y, pz = blockIdx.z;
  const int n0 = nblk * 128;
  const int tid = threadIdx.x, wid = tid >> 5, lane = tid & 31;
  const int lm = lane & 15, g2 = lane >> 4;

  const float* w    = (pz == 0) ? w_g : (pz == 1) ? w_t : w_p;
  const float* bias = (pz == 0) ? b_g : (pz == 1) ? b_t : b_p;
  _Float16* PT = ws + (size_t)pz * ((size_t)Bq * Nn * CI);

  __shared__ __align__(16) _Float16 Xs[128][32];  // [n][k]  8 KB
  __shared__ __align__(16) _Float16 Ws[256][32];  // [o][k] 16 KB

  v8f acc[2][8];
#pragma unroll
  for (int i = 0; i < 2; ++i)
#pragma unroll
    for (int j = 0; j < 8; ++j) acc[i][j] = zero8();

  for (int k0 = 0; k0 < Cc; k0 += 32) {
#pragma unroll
    for (int i = 0; i < 16; ++i) {            // X chunk [32 k x 128 n] -> Xs[n][k]
      int idx = tid + i * 256;
      int cc = idx >> 7, nn = idx & 127;
      Xs[nn][cc] = (_Float16)x[((size_t)b * Cc + k0 + cc) * Nn + n0 + nn];
    }
#pragma unroll
    for (int i = 0; i < 32; ++i) {            // W chunk [256 o x 32 k]
      int idx = tid + i * 256;
      int o = idx >> 5, kk = idx & 31;
      Ws[o][kk] = (_Float16)w[(size_t)o * Cc + k0 + kk];
    }
    __syncthreads();
#pragma unroll
    for (int i2 = 0; i2 < 2; ++i2) {
      const int o = 16 * (2 * wid + i2) + lm;
      v16h Aw = cat16(*(const h8*)&Ws[o][g2 * 8], *(const h8*)&Ws[o][16 + g2 * 8]);
#pragma unroll
      for (int nt = 0; nt < 8; ++nt) {
        v16h Bx = cat16(*(const h8*)&Xs[nt * 16 + lm][g2 * 8],
                        *(const h8*)&Xs[nt * 16 + lm][16 + g2 * 8]);
        acc[i2][nt] = WMMA(Aw, Bx, acc[i2][nt]);
      }
    }
    __syncthreads();
  }

  // epilogue: +bias, store transposed PT[b][n][ci] (one b128 store per tile/lane)
#pragma unroll
  for (int i2 = 0; i2 < 2; ++i2) {
    const int cb = 16 * (2 * wid + i2) + g2 * 8;
    float bb[8];
#pragma unroll
    for (int v = 0; v < 8; ++v) bb[v] = bias[cb + v];
#pragma unroll
    for (int nt = 0; nt < 8; ++nt) {
      h8 hv;
#pragma unroll
      for (int v = 0; v < 8; ++v) hv[v] = (_Float16)(acc[i2][nt][v] + bb[v]);
      const size_t n = (size_t)(n0 + nt * 16 + lm);
      *(h8*)&PT[((size_t)b * Nn + n) * CI + cb] = hv;
    }
  }
}

// ---------------------------------------------------------------------------
// Kernel 2: fused attention.  For a 128-wide n block:
//   loop m (32-chunk): F = theta^T phi / N (regs->LDS f16), Y += g * F^T.
// phi chunk staged via async global->LDS B128 (ASYNCcnt).
// Y stored transposed YT[b][n][ci] f16.  grid (8 nblk, 32 b), block 256.
// ---------------------------------------------------------------------------
__global__ void __launch_bounds__(256)
nl_attn_kernel(const _Float16* __restrict__ ws_in, _Float16* __restrict__ YT) {
  const int nblk = blockIdx.x, b = blockIdx.y;
  const int n0 = nblk * 128;
  const int tid = threadIdx.x, wid = tid >> 5, lane = tid & 31;
  const int lm = lane & 15, g2 = lane >> 4;
  const size_t PN = (size_t)Bq * Nn * CI;
  const _Float16* gT = ws_in;            // [b][n][ci]
  const _Float16* tT = ws_in + PN;       // thetaT
  const _Float16* pT = ws_in + 2 * PN;   // phiT
  const float INVN = 1.0f / (float)Nn;

  __shared__ __align__(16) _Float16 phs[32][256];  // [m][ci] 16 KB
  __shared__ __align__(16) _Float16 gs[256][32];   // [c][m]  16 KB
  __shared__ __align__(16) _Float16 Fs[128][32];   // [n][m]   8 KB

  const _Float16* pbase = pT + (size_t)b * Nn * CI;   // uniform SGPR base

  v8f acc[2][8];
#pragma unroll
  for (int i = 0; i < 2; ++i)
#pragma unroll
    for (int j = 0; j < 8; ++j) acc[i][j] = zero8();

  for (int m0 = 0; m0 < Nn; m0 += 32) {
    // phi chunk: async global->LDS, 16 bytes per lane per issue (no VGPR data)
#pragma unroll
    for (int i = 0; i < 4; ++i) {
      int idx = tid + i * 256;
      int mm = idx >> 5, cg = idx & 31;
      unsigned ldsoff = (unsigned)(size_t)&phs[mm][cg * 8];
      unsigned goff = (unsigned)(((unsigned)(m0 + mm) * CI + cg * 8) * 2u);
      asm volatile("global_load_async_to_lds_b128 %0, %1, %2"
                   :: "v"(ldsoff), "v"(goff), "s"(pbase) : "memory");
    }
#pragma unroll
    for (int i = 0; i < 4; ++i) {          // g chunk: gT rows transposed -> gs[c][m]
      int idx = tid + i * 256;
      int mm = idx >> 5, cg = idx & 31;
      h8 v = *(const h8*)&gT[((size_t)b * Nn + m0 + mm) * CI + cg * 8];
#pragma unroll
      for (int e = 0; e < 8; ++e) gs[cg * 8 + e][mm] = v[e];
    }
    asm volatile("s_wait_asynccnt 0x0" ::: "memory");
    __syncthreads();

    // GEMM1: each wave makes one 16x32 F tile pair (n-tile = wid, m-tiles 0,1)
    v8f facc[2] = {zero8(), zero8()};
    for (int k = 0; k < CI; k += 32) {
      const _Float16* ta = tT + ((size_t)b * Nn + n0 + wid * 16 + lm) * CI + k;
      v16h A = cat16(*(const h8*)(ta + g2 * 8), *(const h8*)(ta + 16 + g2 * 8));
#pragma unroll
      for (int mt = 0; mt < 2; ++mt) {
        v16h Bv = cat16(*(const h8*)&phs[mt * 16 + lm][k + g2 * 8],
                        *(const h8*)&phs[mt * 16 + lm][k + 16 + g2 * 8]);
        facc[mt] = WMMA(A, Bv, facc[mt]);
      }
    }
#pragma unroll
    for (int mt = 0; mt < 2; ++mt)
#pragma unroll
      for (int v = 0; v < 8; ++v)
        Fs[wid * 16 + g2 * 8 + v][mt * 16 + lm] = (_Float16)(facc[mt][v] * INVN);
    __syncthreads();

    // GEMM2: Y[c-tile, n-tile] += g[c,32m] * F^T[32m, n]
#pragma unroll
    for (int i2 = 0; i2 < 2; ++i2) {
      const int c = 16 * (2 * wid + i2) + lm;
      v16h Ag = cat16(*(const h8*)&gs[c][g2 * 8], *(const h8*)&gs[c][16 + g2 * 8]);
#pragma unroll
      for (int nt = 0; nt < 8; ++nt) {
        v16h Bf = cat16(*(const h8*)&Fs[nt * 16 + lm][g2 * 8],
                        *(const h8*)&Fs[nt * 16 + lm][16 + g2 * 8]);
        acc[i2][nt] = WMMA(Ag, Bf, acc[i2][nt]);
      }
    }
    __syncthreads();
  }

#pragma unroll
  for (int i2 = 0; i2 < 2; ++i2) {
    const int cb = 16 * (2 * wid + i2) + g2 * 8;
#pragma unroll
    for (int nt = 0; nt < 8; ++nt) {
      h8 hv;
#pragma unroll
      for (int v = 0; v < 8; ++v) hv[v] = (_Float16)acc[i2][nt][v];
      const size_t n = (size_t)(n0 + nt * 16 + lm);
      *(h8*)&YT[((size_t)b * Nn + n) * CI + cb] = hv;
    }
  }
}

// ---------------------------------------------------------------------------
// Kernel 3: Z = BN(w_W * Y + b_W) + x.  grid (16 nblk of 64, 32 b), block 256.
// Y B-operands load straight from YT[b][n][ci] in WMMA B layout (2 x b128).
// ---------------------------------------------------------------------------
__global__ void __launch_bounds__(256)
nl_out_kernel(const _Float16* __restrict__ YT,
              const float* __restrict__ wW, const float* __restrict__ bW,
              const float* __restrict__ gamma, const float* __restrict__ beta,
              const float* __restrict__ mean, const float* __restrict__ var,
              const float* __restrict__ x, float* __restrict__ out) {
  const int nblk = blockIdx.x, b = blockIdx.y;
  const int n0 = nblk * 64;
  const int tid = threadIdx.x, wid = tid >> 5, lane = tid & 31;
  const int lm = lane & 15, g2 = lane >> 4;

  __shared__ __align__(16) _Float16 Ws[512][32];  // 32 KB

  v8f acc[4][4];
#pragma unroll
  for (int i = 0; i < 4; ++i)
#pragma unroll
    for (int j = 0; j < 4; ++j) acc[i][j] = zero8();

  for (int k0 = 0; k0 < CI; k0 += 32) {
#pragma unroll
    for (int i = 0; i < 64; ++i) {
      int idx = tid + i * 256;
      int o = idx >> 5, kk = idx & 31;
      Ws[o][kk] = (_Float16)wW[(size_t)o * CI + k0 + kk];
    }
    __syncthreads();
    v16h By[4];
#pragma unroll
    for (int nt = 0; nt < 4; ++nt) {
      const _Float16* yp =
          YT + ((size_t)b * Nn + n0 + nt * 16 + lm) * CI + k0 + g2 * 8;
      By[nt] = cat16(*(const h8*)yp, *(const h8*)(yp + 16));
    }
#pragma unroll
    for (int i2 = 0; i2 < 4; ++i2) {
      const int c = 16 * (4 * wid + i2) + lm;
      v16h Aw = cat16(*(const h8*)&Ws[c][g2 * 8], *(const h8*)&Ws[c][16 + g2 * 8]);
#pragma unroll
      for (int nt = 0; nt < 4; ++nt) acc[i2][nt] = WMMA(Aw, By[nt], acc[i2][nt]);
    }
    __syncthreads();
  }

  // fused bias + BN + residual, fp32 out
#pragma unroll
  for (int i2 = 0; i2 < 4; ++i2) {
    const int cb = 16 * (4 * wid + i2) + g2 * 8;
    float s8[8], t8[8];
#pragma unroll
    for (int v = 0; v < 8; ++v) {
      const int c = cb + v;
      const float s = rsqrtf(var[c] + EPSC) * gamma[c];
      s8[v] = s;
      t8[v] = (bW[c] - mean[c]) * s + beta[c];
    }
#pragma unroll
    for (int nt = 0; nt < 4; ++nt) {
      const int n = n0 + nt * 16 + lm;
#pragma unroll
      for (int v = 0; v < 8; ++v) {
        const size_t idx = ((size_t)b * Cc + cb + v) * Nn + n;
        out[idx] = acc[i2][nt][v] * s8[v] + t8[v] + x[idx];
      }
    }
  }
}

// ---------------------------------------------------------------------------
extern "C" void kernel_launch(void* const* d_in, const int* in_sizes, int n_in,
                              void* d_out, int out_size, void* d_ws, size_t ws_size,
                              hipStream_t stream) {
  const float* x     = (const float*)d_in[0];
  const float* w_g   = (const float*)d_in[1];
  const float* b_g   = (const float*)d_in[2];
  const float* w_t   = (const float*)d_in[3];
  const float* b_t   = (const float*)d_in[4];
  const float* w_p   = (const float*)d_in[5];
  const float* b_p   = (const float*)d_in[6];
  const float* w_W   = (const float*)d_in[7];
  const float* b_W   = (const float*)d_in[8];
  const float* gamma = (const float*)d_in[9];
  const float* beta  = (const float*)d_in[10];
  const float* mean  = (const float*)d_in[11];
  const float* var   = (const float*)d_in[12];
  float* out = (float*)d_out;

  _Float16* ws = (_Float16*)d_ws;                    // gT | thetaT | phiT | YT
  const size_t PN = (size_t)Bq * Nn * CI;
  _Float16* YT = ws + 3 * PN;                        // needs 64 MB total

  dim3 g1(Nn / 128, Bq, 3);
  nl_proj_kernel<<<g1, 256, 0, stream>>>(x, w_g, b_g, w_t, b_t, w_p, b_p, ws);

  dim3 g2(Nn / 128, Bq);
  nl_attn_kernel<<<g2, 256, 0, stream>>>(ws, YT);

  dim3 g3(Nn / 64, Bq);
  nl_out_kernel<<<g3, 256, 0, stream>>>(YT, w_W, b_W, gamma, beta, mean, var, x, out);
}
